// CustomRNN_53420803228164
// MI455X (gfx1250) — matrix operand order repaired and checked by reading, not compiled, not run on hardware
//
#include <hip/hip_runtime.h>
#include <hip/hip_bf16.h>
#include <stdint.h>
#include <stddef.h>

#define Bn 32
#define Tn 256
#define Ln 196
#define Vn 8192
#define En 512
#define Un 1024
#define Dn 1024

typedef __attribute__((ext_vector_type(16))) __bf16 v16bf;
typedef __attribute__((ext_vector_type(8)))  __bf16 v8bf;
typedef __attribute__((ext_vector_type(8)))  float  v8f;

// ---- helpers -------------------------------------------------------------

__device__ __forceinline__ __bf16 f2bf(float f) {
  union { float f; uint32_t u; } x; x.f = f;
  uint32_t r = x.u + 0x7FFFu + ((x.u >> 16) & 1u);   // round-to-nearest-even
  uint16_t h = (uint16_t)(r >> 16);
  __bf16 o; __builtin_memcpy(&o, &h, 2); return o;
}

// A fragment (16x32 bf16, row-major source, lda in elements).
// ISA layout: lanes 0-15 row M=lane, elems 0-7 -> K 0..7, elems 8-15 -> K 16..23;
// lanes 16-31 same rows, K shifted by +8.
__device__ __forceinline__ v16bf load_a(const __bf16* A, int lda, int m0, int k0, int lane) {
  const __bf16* p = A + (size_t)(m0 + (lane & 15)) * lda + k0 + ((lane >> 4) << 3);
  union { v16bf v; v8bf h[2]; } u;
  u.h[0] = *(const v8bf*)(p);
  u.h[1] = *(const v8bf*)(p + 16);
  return u.v;
}

// B fragment from pre-packed weights: packed[((nt*Ktiles + kt)*32 + lane)*16 + e]
__device__ __forceinline__ v16bf load_b(const __bf16* Bp, int Ktiles, int kt, int nt, int lane) {
  const __bf16* p = Bp + ((((size_t)nt * Ktiles + kt) * 32) + lane) * 16;
  union { v16bf v; v8bf h[2]; } u;
  u.h[0] = *(const v8bf*)(p);
  u.h[1] = *(const v8bf*)(p + 8);
  return u.v;
}

__device__ __forceinline__ v8f wmma_bf(v16bf a, v16bf b, v8f c) {
  return __builtin_amdgcn_wmma_f32_16x16x32_bf16(false, a, false, b, (short)0, c, false, false);
}

// ---- prep kernels --------------------------------------------------------

// Pack fp32 [K,N] row-major weight into WMMA-B bf16 fragment order.
// B layout: lane holds N = lane&15; elem e -> K = (e&7) + 16*(e>=8) + 8*(lane>=16).
__global__ void k_pack_b(const float* __restrict__ W, __bf16* __restrict__ out, int K, int N) {
  size_t id = (size_t)blockIdx.x * 256 + threadIdx.x;
  size_t tot = (size_t)K * N;
  if (id >= tot) return;
  int e = (int)(id & 15);
  int lane = (int)((id >> 4) & 31);
  size_t tile = id >> 9;
  int Ktiles = K >> 5;
  int kt = (int)(tile % (size_t)Ktiles);
  int nt = (int)(tile / (size_t)Ktiles);
  int n = nt * 16 + (lane & 15);
  int k = kt * 32 + (e & 7) + ((e >> 3) << 4) + ((lane >> 4) << 3);
  out[id] = f2bf(W[(size_t)k * N + n]);
}

__global__ void k_cvt(const float* __restrict__ s, __bf16* __restrict__ d, size_t n) {
  size_t i = (size_t)blockIdx.x * 256 + threadIdx.x;
  if (i < n) d[i] = f2bf(s[i]);
}

__global__ void k_zero32(uint32_t* __restrict__ p, size_t n) {
  size_t i = (size_t)blockIdx.x * 256 + threadIdx.x;
  if (i < n) p[i] = 0u;
}

// Embedding gather -> xg[t][b][e] (bf16), time-major so each step is contiguous.
__global__ void k_gather(const int* __restrict__ formula, const float* __restrict__ emb,
                         __bf16* __restrict__ xg) {
  int id = blockIdx.x;                 // 0..B*T-1
  int b = id / Tn, t = id % Tn;
  int v = formula[(size_t)b * Tn + t];
  const float* src = emb + (size_t)v * En;
  __bf16* dst = xg + ((size_t)t * Bn + b) * En;
  for (int e = threadIdx.x; e < En; e += 256) dst[e] = f2bf(src[e]);
}

// keys = memory @ W1 + b1   (fp32 out [B*L, U]);  grid (392, 8), 256 thr (8 waves)
__global__ void k_keys(const __bf16* __restrict__ membf, const __bf16* __restrict__ W1p,
                       const float* __restrict__ b1, float* __restrict__ keys) {
  int tid = threadIdx.x, lane = tid & 31, w = tid >> 5;
  int m0 = blockIdx.x * 16;
  int nt = blockIdx.y * 8 + w;
  v8f acc = {};
  for (int kt = 0; kt < 32; ++kt) {
    v16bf a = load_a(membf, Dn, m0, kt * 32, lane);
    acc = wmma_bf(a, load_b(W1p, 32, kt, nt, lane), acc);
  }
  int n = nt * 16 + (lane & 15), hi = lane >> 4;
  float bias = b1[n];
#pragma unroll
  for (int j = 0; j < 8; ++j) {
    int m = m0 + j + 8 * hi;
    keys[(size_t)m * Un + n] = acc[j] + bias;
  }
}

// ---- per-step kernels ----------------------------------------------------

// GRU step: h_new from x_t, o_prev, h_prev. grid 16 blocks x 256 thr.
// Block covers 64 output cols; wave w: m-tile = w>>2, n-tile = w&3.
__global__ void k_gru(const __bf16* __restrict__ xt,     // [B,E] (time slice)
                      const __bf16* __restrict__ oprev,  // [B,U]
                      const __bf16* __restrict__ hprevb, // [B,U] bf16
                      const float*  __restrict__ hprevf, // [B,U] fp32
                      const __bf16* __restrict__ Wkp,    // packed [1536,3072]
                      const __bf16* __restrict__ Wrp,    // packed [1024,3072]
                      const float* __restrict__ bin, const float* __restrict__ brec,
                      float* __restrict__ hnewf, __bf16* __restrict__ hnewb) {
  int tid = threadIdx.x, lane = tid & 31, w = tid >> 5;
  int c0 = blockIdx.x * 64;
  int m0 = (w >> 2) * 16;
  int col0 = c0 + (w & 3) * 16;
  int ntz = col0 >> 4, ntr = ntz + 64, nth = ntz + 128;  // z / r / h thirds of N=3072
  v8f az = {}, ar = {}, axh = {}, ahh = {};
  for (int kt = 0; kt < 16; ++kt) {          // K 0..511: x_t part of ci
    v16bf a = load_a(xt, En, m0, kt * 32, lane);
    az  = wmma_bf(a, load_b(Wkp, 48, kt, ntz, lane), az);
    ar  = wmma_bf(a, load_b(Wkp, 48, kt, ntr, lane), ar);
    axh = wmma_bf(a, load_b(Wkp, 48, kt, nth, lane), axh);
  }
  for (int kt = 16; kt < 48; ++kt) {         // K 512..1535: o_prev part of ci
    v16bf a = load_a(oprev, Un, m0, (kt - 16) * 32, lane);
    az  = wmma_bf(a, load_b(Wkp, 48, kt, ntz, lane), az);
    ar  = wmma_bf(a, load_b(Wkp, 48, kt, ntr, lane), ar);
    axh = wmma_bf(a, load_b(Wkp, 48, kt, nth, lane), axh);
  }
  for (int kt = 0; kt < 32; ++kt) {          // gh = h_prev @ Wr
    v16bf a = load_a(hprevb, Un, m0, kt * 32, lane);
    az  = wmma_bf(a, load_b(Wrp, 32, kt, ntz, lane), az);
    ar  = wmma_bf(a, load_b(Wrp, 32, kt, ntr, lane), ar);
    ahh = wmma_bf(a, load_b(Wrp, 32, kt, nth, lane), ahh);
  }
  int n = col0 + (lane & 15), hi = lane >> 4;
  float bz  = bin[n] + brec[n];
  float br  = bin[Un + n] + brec[Un + n];
  float bxh = bin[2 * Un + n], bhh = brec[2 * Un + n];
#pragma unroll
  for (int j = 0; j < 8; ++j) {
    int m = m0 + j + 8 * hi;
    float z  = 1.0f / (1.0f + expf(-(az[j] + bz)));
    float r  = 1.0f / (1.0f + expf(-(ar[j] + br)));
    float hc = tanhf(axh[j] + bxh + r * (ahh[j] + bhh));
    float hp = hprevf[(size_t)m * Un + n];
    float hn = z * hp + (1.0f - z) * hc;
    hnewf[(size_t)m * Un + n] = hn;
    hnewb[(size_t)m * Un + n] = f2bf(hn);
  }
}

// q = h_new @ W2 + b2 (fp32). grid 16 blocks x 256 thr (2 m-tiles x 4 n-tiles / block)
__global__ void k_qproj(const __bf16* __restrict__ h, const __bf16* __restrict__ W2p,
                        const float* __restrict__ b2, float* __restrict__ q) {
  int tid = threadIdx.x, lane = tid & 31, w = tid >> 5;
  int m0 = (w & 1) * 16;
  int nt = blockIdx.x * 4 + (w >> 1);
  v8f acc = {};
  for (int kt = 0; kt < 32; ++kt) {
    v16bf a = load_a(h, Un, m0, kt * 32, lane);
    acc = wmma_bf(a, load_b(W2p, 32, kt, nt, lane), acc);
  }
  int n = nt * 16 + (lane & 15), hi = lane >> 4;
  float bias = b2[n];
#pragma unroll
  for (int j = 0; j < 8; ++j) {
    int m = m0 + j + 8 * hi;
    q[(size_t)m * Un + n] = acc[j] + bias;
  }
}

// Bahdanau attention + context: one block per batch row. 32 blocks x 256 thr.
__global__ void k_attention(const float* __restrict__ q, const float* __restrict__ keys,
                            const float* __restrict__ memory, const float* __restrict__ vatt,
                            __bf16* __restrict__ ctxbf) {
  __shared__ float qs[Un];
  __shared__ float sc[224];
  int b = blockIdx.x;
  int tid = threadIdx.x, lane = tid & 31, w = tid >> 5;
  for (int i = tid; i < Un; i += 256) qs[i] = q[(size_t)b * Un + i];
  __syncthreads();
  for (int l = w; l < Ln; l += 8) {
    const float* kp = keys + ((size_t)b * Ln + l) * Un;
    float s = 0.0f;
    for (int u = lane; u < Un; u += 32) s += tanhf(kp[u] + qs[u]) * vatt[u];
#pragma unroll
    for (int off = 16; off; off >>= 1) s += __shfl_xor(s, off, 32);
    if (lane == 0) sc[l] = s;
  }
  __syncthreads();
  if (w == 0) {                               // softmax over L=196 by one wave
    float mx = -1e30f;
    for (int l = lane; l < Ln; l += 32) mx = fmaxf(mx, sc[l]);
#pragma unroll
    for (int off = 16; off; off >>= 1) mx = fmaxf(mx, __shfl_xor(mx, off, 32));
    float sum = 0.0f;
    for (int l = lane; l < Ln; l += 32) { float e = expf(sc[l] - mx); sc[l] = e; sum += e; }
#pragma unroll
    for (int off = 16; off; off >>= 1) sum += __shfl_xor(sum, off, 32);
    float inv = 1.0f / sum;
    for (int l = lane; l < Ln; l += 32) sc[l] *= inv;
  }
  __syncthreads();
  for (int d = tid; d < Dn; d += 256) {       // ctx = w @ memory[b]
    float acc = 0.0f;
    const float* mp = memory + (size_t)b * Ln * Dn + d;
    for (int l = 0; l < Ln; ++l) acc += sc[l] * mp[(size_t)l * Dn];
    ctxbf[(size_t)b * Dn + d] = f2bf(acc);
  }
}

// o_new = tanh(concat(ctx, h_new) @ Wout). grid 16 x 256.
__global__ void k_outproj(const __bf16* __restrict__ ctx, const __bf16* __restrict__ h,
                          const __bf16* __restrict__ Wop, __bf16* __restrict__ onew,
                          __bf16* __restrict__ outs, int t) {
  int tid = threadIdx.x, lane = tid & 31, w = tid >> 5;
  int m0 = (w & 1) * 16;
  int nt = blockIdx.x * 4 + (w >> 1);
  v8f acc = {};
  for (int kt = 0; kt < 32; ++kt) {
    v16bf a = load_a(ctx, Dn, m0, kt * 32, lane);
    acc = wmma_bf(a, load_b(Wop, 64, kt, nt, lane), acc);
  }
  for (int kt = 32; kt < 64; ++kt) {
    v16bf a = load_a(h, Un, m0, (kt - 32) * 32, lane);
    acc = wmma_bf(a, load_b(Wop, 64, kt, nt, lane), acc);
  }
  int n = nt * 16 + (lane & 15), hi = lane >> 4;
#pragma unroll
  for (int j = 0; j < 8; ++j) {
    int m = m0 + j + 8 * hi;
    __bf16 ob = f2bf(tanhf(acc[j]));
    onew[(size_t)m * Un + n] = ob;
    outs[((size_t)m * Tn + t) * Un + n] = ob;     // row b*T+t of outs
  }
}

// logits = outs @ Wproj, fp32 out. grid (256, 32) x 256 thr.
// Block tile: 32 rows x 256 cols; wave: 1 m-tile, 4 n-tiles (A fragment reuse).
__global__ void k_vocab(const __bf16* __restrict__ outs, const __bf16* __restrict__ Wp,
                        float* __restrict__ out) {
  int tid = threadIdx.x, lane = tid & 31, w = tid >> 5;
  int m0 = blockIdx.x * 32 + (w & 1) * 16;
  int n0 = blockIdx.y * 256 + (w >> 1) * 64;
  int ntb = n0 >> 4;
  v8f a0 = {}, a1 = {}, a2 = {}, a3 = {};
  for (int kt = 0; kt < 32; ++kt) {
    v16bf a = load_a(outs, Un, m0, kt * 32, lane);
    a0 = wmma_bf(a, load_b(Wp, 32, kt, ntb + 0, lane), a0);
    a1 = wmma_bf(a, load_b(Wp, 32, kt, ntb + 1, lane), a1);
    a2 = wmma_bf(a, load_b(Wp, 32, kt, ntb + 2, lane), a2);
    a3 = wmma_bf(a, load_b(Wp, 32, kt, ntb + 3, lane), a3);
  }
  int hi = lane >> 4, nl = lane & 15;
#pragma unroll
  for (int j = 0; j < 8; ++j) {
    int m = m0 + j + 8 * hi;
    float* row = out + (size_t)m * Vn;
    row[n0 +  0 + nl] = a0[j];
    row[n0 + 16 + nl] = a1[j];
    row[n0 + 32 + nl] = a2[j];
    row[n0 + 48 + nl] = a3[j];
  }
}

// ---- host orchestration --------------------------------------------------

extern "C" void kernel_launch(void* const* d_in, const int* in_sizes, int n_in,
                              void* d_out, int out_size, void* d_ws, size_t ws_size,
                              hipStream_t stream) {
  (void)in_sizes; (void)n_in; (void)out_size; (void)ws_size;
  const int*   formula = (const int*)  d_in[0];
  const float* memory  = (const float*)d_in[1];
  const float* emb     = (const float*)d_in[2];
  const float* gk      = (const float*)d_in[3];
  const float* grk     = (const float*)d_in[4];
  const float* gbi     = (const float*)d_in[5];
  const float* gbr     = (const float*)d_in[6];
  const float* W1      = (const float*)d_in[7];
  const float* b1      = (const float*)d_in[8];
  const float* W2      = (const float*)d_in[9];
  const float* b2      = (const float*)d_in[10];
  const float* vatt    = (const float*)d_in[11];
  const float* Wout    = (const float*)d_in[12];
  const float* Wproj   = (const float*)d_in[13];
  float* out = (float*)d_out;

  char* p = (char*)d_ws;
  auto carve = [&](size_t bytes) -> void* {
    void* r = (void*)p; p += (bytes + 255) & ~(size_t)255; return r;
  };
  __bf16* Wkp    = (__bf16*)carve((size_t)1536 * 3072 * 2);
  __bf16* Wrp    = (__bf16*)carve((size_t)1024 * 3072 * 2);
  __bf16* W1p    = (__bf16*)carve((size_t)1024 * 1024 * 2);
  __bf16* W2p    = (__bf16*)carve((size_t)1024 * 1024 * 2);
  __bf16* Woutp  = (__bf16*)carve((size_t)2048 * 1024 * 2);
  __bf16* Wprojp = (__bf16*)carve((size_t)1024 * 8192 * 2);
  __bf16* membf  = (__bf16*)carve((size_t)Bn * Ln * Dn * 2);
  __bf16* xg     = (__bf16*)carve((size_t)Tn * Bn * En * 2);
  float*  keys   = (float*) carve((size_t)Bn * Ln * Un * 4);
  __bf16* outsb  = (__bf16*)carve((size_t)Bn * Tn * Un * 2);
  float*  hf0    = (float*) carve((size_t)Bn * Un * 4);
  float*  hf1    = (float*) carve((size_t)Bn * Un * 4);
  __bf16* hb0    = (__bf16*)carve((size_t)Bn * Un * 2);
  __bf16* hb1    = (__bf16*)carve((size_t)Bn * Un * 2);
  __bf16* ob0    = (__bf16*)carve((size_t)Bn * Un * 2);
  __bf16* ob1    = (__bf16*)carve((size_t)Bn * Un * 2);
  float*  q      = (float*) carve((size_t)Bn * Un * 4);
  __bf16* ctxbf  = (__bf16*)carve((size_t)Bn * Dn * 2);

  float*  hf[2] = { hf0, hf1 };
  __bf16* hb[2] = { hb0, hb1 };
  __bf16* ob[2] = { ob0, ob1 };

  auto blocks = [](size_t n) { return (unsigned)((n + 255) / 256); };

  // Weight packing (bf16 fragment layout)
  k_pack_b<<<blocks((size_t)1536 * 3072), 256, 0, stream>>>(gk,    Wkp,    1536, 3072);
  k_pack_b<<<blocks((size_t)1024 * 3072), 256, 0, stream>>>(grk,   Wrp,    1024, 3072);
  k_pack_b<<<blocks((size_t)1024 * 1024), 256, 0, stream>>>(W1,    W1p,    1024, 1024);
  k_pack_b<<<blocks((size_t)1024 * 1024), 256, 0, stream>>>(W2,    W2p,    1024, 1024);
  k_pack_b<<<blocks((size_t)2048 * 1024), 256, 0, stream>>>(Wout,  Woutp,  2048, 1024);
  k_pack_b<<<blocks((size_t)1024 * 8192), 256, 0, stream>>>(Wproj, Wprojp, 1024, 8192);

  size_t memn = (size_t)Bn * Ln * Dn;
  k_cvt<<<blocks(memn), 256, 0, stream>>>(memory, membf, memn);
  k_gather<<<Bn * Tn, 256, 0, stream>>>(formula, emb, xg);

  // zero initial h (fp32 + bf16) and o (bf16)
  k_zero32<<<blocks((size_t)Bn * Un), 256, 0, stream>>>((uint32_t*)hf[0], (size_t)Bn * Un);
  k_zero32<<<blocks((size_t)Bn * Un / 2), 256, 0, stream>>>((uint32_t*)hb[0], (size_t)Bn * Un / 2);
  k_zero32<<<blocks((size_t)Bn * Un / 2), 256, 0, stream>>>((uint32_t*)ob[0], (size_t)Bn * Un / 2);

  // keys = memory @ W1 + b1
  k_keys<<<dim3((Bn * Ln) / 16, 8), 256, 0, stream>>>(membf, W1p, b1, keys);

  // sequential scan
  for (int t = 0; t < Tn; ++t) {
    int cur = t & 1, nxt = cur ^ 1;
    const __bf16* xt = xg + (size_t)t * Bn * En;
    k_gru<<<16, 256, 0, stream>>>(xt, ob[cur], hb[cur], hf[cur], Wkp, Wrp,
                                  gbi, gbr, hf[nxt], hb[nxt]);
    k_qproj<<<16, 256, 0, stream>>>(hb[nxt], W2p, b2, q);
    k_attention<<<Bn, 256, 0, stream>>>(q, keys, memory, vatt, ctxbf);
    k_outproj<<<16, 256, 0, stream>>>(ctxbf, hb[nxt], Woutp, ob[nxt], outsb, t);
  }

  // logits = outs @ Wproj
  k_vocab<<<dim3(256, 32), 256, 0, stream>>>(outsb, Wprojp, out);
}
